// Decoder_62766652064077
// MI455X (gfx1250) — matrix operand order repaired and checked by reading, not compiled, not run on hardware
//
#include <hip/hip_runtime.h>
#include <hip/hip_bf16.h>
#include <cstdint>

// ---------------- problem constants (from reference) ----------------
#define NPED   4096
#define SCN    64
#define PPS    64      // peds per scene
#define HD     64      // h_dim
#define ED     16      // embedding dim
#define PPH    512     // mlp_pre_pool hidden
#define BOT    32
#define MLPD   64
#define PREDL  12

typedef __attribute__((ext_vector_type(16))) _Float16 v16h;
typedef __attribute__((ext_vector_type(8)))  float    v8f;

union AFrag { v16h v; uint32_t u[8]; };
union BFrag { v16h v; uint4 q[2]; };

// ---------------- init helpers ----------------
__global__ void copy_k(const float* __restrict__ src, float* __restrict__ dst, int n) {
    int i = blockIdx.x * blockDim.x + threadIdx.x;
    if (i < n) dst[i] = src[i];
}

// x0 = [last_pos_rel, spd0] @ W_sp + b_sp
__global__ void x0_k(const float* __restrict__ lpr, const float* __restrict__ spd0,
                     const float* __restrict__ W_sp, const float* __restrict__ b_sp,
                     float* __restrict__ x) {
    int idx = blockIdx.x * blockDim.x + threadIdx.x;
    if (idx >= NPED * ED) return;
    int n = idx >> 4, e = idx & 15;
    x[idx] = b_sp[e] + lpr[n*2]*W_sp[e] + lpr[n*2+1]*W_sp[16+e] + spd0[n]*W_sp[32+e];
}

// W3[3,512] = W_pe @ W_pp1[64:80,:] ; cvec[512] = b_pp1 + b_pe @ W_pp1[64:80,:]
__global__ void prep_k(const float* __restrict__ W_pe, const float* __restrict__ b_pe,
                       const float* __restrict__ W_pp1, const float* __restrict__ b_pp1,
                       float* __restrict__ W3, float* __restrict__ cvec) {
    int k = blockIdx.x * blockDim.x + threadIdx.x;
    if (k >= PPH) return;
    for (int d = 0; d < 3; ++d) {
        float a = 0.f;
        for (int e = 0; e < ED; ++e) a += W_pe[d*ED + e] * W_pp1[(HD + e)*PPH + k];
        W3[d*PPH + k] = a;
    }
    float cb = b_pp1[k];
    for (int e = 0; e < ED; ++e) cb += b_pe[e] * W_pp1[(HD + e)*PPH + k];
    cvec[k] = cb;
}

// Pre-pack W_pp2 (f32 [512,32]) into WMMA B-fragments (f16), contiguous-K lane layout (same as A):
// lane n-col = lane&15 ; K halves: lanes 0-15 hold K kb..kb+15, lanes 16-31 K kb+16..kb+31.
// Layout: bfrag[kc(16)][ntile(2)][lane(32)][8 uints]
__global__ void bfrag_k(const float* __restrict__ W_pp2, uint32_t* __restrict__ bfrag) {
    int idx = blockIdx.x * blockDim.x + threadIdx.x;
    if (idx >= 16 * 2 * 32 * 8) return;
    int q    = idx & 7;
    int lane = (idx >> 3) & 31;
    int nt   = (idx >> 8) & 1;
    int kc   = idx >> 9;
    int n    = nt * 16 + (lane & 15);
    int kb   = kc * 32 + ((lane < 16) ? 0 : 16);
    _Float16 h0 = (_Float16)W_pp2[(kb + 2*q    ) * BOT + n];
    _Float16 h1 = (_Float16)W_pp2[(kb + 2*q + 1) * BOT + n];
    uint32_t w = ((uint32_t)(*(unsigned short*)&h1) << 16) | (uint32_t)(*(unsigned short*)&h0);
    bfrag[idx] = w;
}

// ---------------- per-step kernels ----------------
// LSTM cell + hidden2pos + curr + spatial embedding. One block (256 thr) per ped.
__global__ __launch_bounds__(256) void lstm_step(
    const float* __restrict__ W_ih, const float* __restrict__ b_ih,
    const float* __restrict__ W_hh, const float* __restrict__ b_hh,
    const float* __restrict__ W_h2p, const float* __restrict__ b_h2p,
    const float* __restrict__ W_sp, const float* __restrict__ b_sp,
    const float* __restrict__ spd_t,
    float* __restrict__ h, float* __restrict__ c,
    float* __restrict__ lp, float* __restrict__ x,
    float* __restrict__ rel_out) {
    int n = blockIdx.x, t = threadIdx.x;
    __shared__ float xs[ED], hs[HD], g[4*HD], hn[HD], rp[2];
    if (t < ED) xs[t] = x[n*ED + t];
    if (t < HD) hs[t] = h[n*HD + t];
    __syncthreads();
    float acc = b_ih[t] + b_hh[t];
    #pragma unroll 4
    for (int k = 0; k < ED; ++k) acc += xs[k] * W_ih[k*256 + t];
    #pragma unroll 4
    for (int k = 0; k < HD; ++k) acc += hs[k] * W_hh[k*256 + t];
    g[t] = acc;
    __syncthreads();
    if (t < HD) {
        float ig = 1.f / (1.f + __expf(-g[t]));
        float fg = 1.f / (1.f + __expf(-g[HD + t]));
        float gg = tanhf(g[2*HD + t]);
        float og = 1.f / (1.f + __expf(-g[3*HD + t]));
        float cn = fg * c[n*HD + t] + ig * gg;
        float hv = og * tanhf(cn);
        c[n*HD + t] = cn;
        h[n*HD + t] = hv;
        hn[t] = hv;
    }
    __syncthreads();
    if (t < 2) {
        float r = b_h2p[t];
        for (int k = 0; k < HD; ++k) r += hn[k] * W_h2p[k*2 + t];
        rp[t] = r;
        rel_out[n*2 + t] = r;
        lp[n*2 + t] = r + lp[n*2 + t];   // curr position
    }
    __syncthreads();
    if (t < ED) {
        float s = spd_t[n];
        x[n*ED + t] = b_sp[t] + rp[0]*W_sp[t] + rp[1]*W_sp[16 + t] + s*W_sp[32 + t];
    }
}

// u[n,512] = cvec + h[n]@W_pp1[:64] + px*W3[0] + py*W3[1] + spd*W3[2] ; v[n,512] = px*W3[0]+py*W3[1]
__global__ __launch_bounds__(256) void uv_kernel(
    const float* __restrict__ W_pp1, const float* __restrict__ W3,
    const float* __restrict__ cvec, const float* __restrict__ h,
    const float* __restrict__ lp, const float* __restrict__ spd_t,
    float* __restrict__ u, float* __restrict__ v) {
    int n = blockIdx.x, t = threadIdx.x;
    __shared__ float hs[HD];
    if (t < HD) hs[t] = h[n*HD + t];
    __syncthreads();
    float px = lp[n*2], py = lp[n*2 + 1], sp = spd_t[n];
    #pragma unroll
    for (int kk = 0; kk < 2; ++kk) {
        int k = t + kk*256;
        float a = cvec[k];
        #pragma unroll 4
        for (int j = 0; j < HD; ++j) a += hs[j] * W_pp1[j*PPH + k];
        float vv = px * W3[k] + py * W3[PPH + k];
        u[(size_t)n*PPH + k] = a + vv + sp * W3[2*PPH + k];
        v[(size_t)n*PPH + k] = vv;
    }
}

// Pool: one block (128 thr / 4 waves) per (scene s, focal ped i).
// x1[j,k] = relu(u[s,j,k]-v[s,i,k]) (f16) ; y = relu(x1 @ W_pp2 + b_pp2) ; pool = max_j y
__global__ __launch_bounds__(128) void pool_kernel(
    const float* __restrict__ u, const float* __restrict__ v,
    const uint32_t* __restrict__ bfrag, const float* __restrict__ b_pp2,
    float* __restrict__ pool) {
    __shared__ uint32_t x1u[PPS * 16];   // 64 j x 32 k f16, packed pairs
    __shared__ float red[4][BOT];
    int bid = blockIdx.x;
    int s = bid >> 6, i = bid & 63;
    int t = threadIdx.x;
    int wave = t >> 5, lane = t & 31;

    const float* ubase = u + (size_t)(s * PPS) * PPH;
    const float* vrow  = v + (size_t)(s * PPS + i) * PPH;

    int jb = t >> 1;               // build: j row
    int kh = (t & 1) * 16;         // build: 16 consecutive k

    v8f acc0 = {}; v8f acc1 = {};

    for (int kc = 0; kc < 16; ++kc) {
        __syncthreads();           // protect previous iteration's fragment reads
        {   // cooperative x1 chunk build
            const float* up = ubase + (size_t)jb * PPH + kc*32 + kh;
            const float* vp = vrow + kc*32 + kh;
            uint32_t* dst = &x1u[jb*16 + (kh >> 1)];
            #pragma unroll
            for (int q = 0; q < 8; ++q) {
                float a0 = up[2*q]     - vp[2*q];
                float a1 = up[2*q + 1] - vp[2*q + 1];
                a0 = a0 > 0.f ? a0 : 0.f;
                a1 = a1 > 0.f ? a1 : 0.f;
                _Float16 h0 = (_Float16)a0, h1 = (_Float16)a1;
                dst[q] = ((uint32_t)(*(unsigned short*)&h1) << 16) |
                          (uint32_t)(*(unsigned short*)&h0);
            }
        }
        __syncthreads();
        // A fragment, m-tile = wave.  Contiguous K layout (matches bfrag):
        // lanes 0-15: K 0-15 (u[0..7]); lanes 16-31: K 16-31.
        AFrag a;
        int m  = lane & 15;
        int jg = wave * 16 + m;
        int k0h = (lane < 16) ? 0 : 8;          // uint offset = K0/2
        const uint32_t* row = &x1u[jg * 16];
        #pragma unroll
        for (int q = 0; q < 4; ++q) {
            a.u[q]     = row[k0h + q];
            a.u[4 + q] = row[4 + k0h + q];
        }
        // B fragments (pre-packed, coalesced b128 loads)
        BFrag b0, b1;
        const uint4* bq0 = (const uint4*)(bfrag + ((size_t)(kc*2 + 0)*32 + lane)*8);
        const uint4* bq1 = (const uint4*)(bfrag + ((size_t)(kc*2 + 1)*32 + lane)*8);
        b0.q[0] = bq0[0]; b0.q[1] = bq0[1];
        b1.q[0] = bq1[0]; b1.q[1] = bq1[1];

        acc0 = __builtin_amdgcn_wmma_f32_16x16x32_f16(false, a.v, false, b0.v,
                                                      (short)0, acc0, false, false);
        acc1 = __builtin_amdgcn_wmma_f32_16x16x32_f16(false, a.v, false, b1.v,
                                                      (short)0, acc1, false, false);
    }

    // epilogue: bias + relu + max over the 16 j of this wave's m-tile
    int n0 = lane & 15;
    float bias0 = b_pp2[n0], bias1 = b_pp2[16 + n0];
    float m0 = 0.f, m1 = 0.f;               // relu output >= 0
    #pragma unroll
    for (int r = 0; r < 8; ++r) {
        float y0 = acc0[r] + bias0; y0 = y0 > 0.f ? y0 : 0.f;
        float y1 = acc1[r] + bias1; y1 = y1 > 0.f ? y1 : 0.f;
        m0 = fmaxf(m0, y0);
        m1 = fmaxf(m1, y1);
    }
    m0 = fmaxf(m0, __shfl_xor(m0, 16));     // combine M 0-7 with M 8-15 (wave32)
    m1 = fmaxf(m1, __shfl_xor(m1, 16));
    if (lane < 16) { red[wave][n0] = m0; red[wave][16 + n0] = m1; }
    __syncthreads();
    if (t < BOT) {
        float r = fmaxf(fmaxf(red[0][t], red[1][t]), fmaxf(red[2][t], red[3][t]));
        pool[(size_t)(s * PPS + i) * BOT + t] = r;
    }
}

// Decoder MLP: h = relu(relu([h,pool] @ W_m1 + b_m1) @ W_m2 + b_m2). 64 thr per ped.
__global__ __launch_bounds__(64) void mlp_kernel(
    const float* __restrict__ W_m1, const float* __restrict__ b_m1,
    const float* __restrict__ W_m2, const float* __restrict__ b_m2,
    const float* __restrict__ pool, float* __restrict__ h) {
    int n = blockIdx.x, t = threadIdx.x;
    __shared__ float in96[HD + BOT], hid[MLPD];
    in96[t] = h[n*HD + t];
    if (t < BOT) in96[HD + t] = pool[n*BOT + t];
    __syncthreads();
    float a = b_m1[t];
    #pragma unroll 4
    for (int k = 0; k < HD + BOT; ++k) a += in96[k] * W_m1[k*MLPD + t];
    hid[t] = a > 0.f ? a : 0.f;
    __syncthreads();
    float o = b_m2[t];
    #pragma unroll 4
    for (int k = 0; k < MLPD; ++k) o += hid[k] * W_m2[k*HD + t];
    h[n*HD + t] = o > 0.f ? o : 0.f;
}

// ---------------- launcher ----------------
extern "C" void kernel_launch(void* const* d_in, const int* in_sizes, int n_in,
                              void* d_out, int out_size, void* d_ws, size_t ws_size,
                              hipStream_t stream) {
    const float* last_pos     = (const float*)d_in[0];
    const float* last_pos_rel = (const float*)d_in[1];
    const float* h0   = (const float*)d_in[2];
    const float* c0   = (const float*)d_in[3];
    const float* spd  = (const float*)d_in[4];   // [12, N, 1]
    const float* W_sp = (const float*)d_in[5];   const float* b_sp = (const float*)d_in[6];
    const float* W_ih = (const float*)d_in[7];   const float* b_ih = (const float*)d_in[8];
    const float* W_hh = (const float*)d_in[9];   const float* b_hh = (const float*)d_in[10];
    const float* W_h2p= (const float*)d_in[11];  const float* b_h2p= (const float*)d_in[12];
    const float* W_pe = (const float*)d_in[13];  const float* b_pe = (const float*)d_in[14];
    const float* W_pp1= (const float*)d_in[15];  const float* b_pp1= (const float*)d_in[16];
    const float* W_pp2= (const float*)d_in[17];  const float* b_pp2= (const float*)d_in[18];
    const float* W_m1 = (const float*)d_in[19];  const float* b_m1 = (const float*)d_in[20];
    const float* W_m2 = (const float*)d_in[21];  const float* b_m2 = (const float*)d_in[22];
    float* out = (float*)d_out;

    // workspace layout (256B aligned)
    char* ws = (char*)d_ws;
    size_t off = 0;
    auto alloc = [&](size_t bytes) { char* p = ws + off; off = (off + bytes + 255) & ~(size_t)255; return p; };
    float*    h     = (float*)alloc((size_t)NPED*HD*4);
    float*    c     = (float*)alloc((size_t)NPED*HD*4);
    float*    lp    = (float*)alloc((size_t)NPED*2*4);
    float*    x     = (float*)alloc((size_t)NPED*ED*4);
    float*    u     = (float*)alloc((size_t)NPED*PPH*4);
    float*    v     = (float*)alloc((size_t)NPED*PPH*4);
    float*    pool  = (float*)alloc((size_t)NPED*BOT*4);
    float*    W3    = (float*)alloc(3*PPH*4);
    float*    cvec  = (float*)alloc(PPH*4);
    uint32_t* bfrag = (uint32_t*)alloc(16*2*32*8*4);
    (void)in_sizes; (void)n_in; (void)out_size; (void)ws_size;

    // ---- init (state copies + folded constants) ----
    copy_k<<<(NPED*HD + 255)/256, 256, 0, stream>>>(h0, h, NPED*HD);
    copy_k<<<(NPED*HD + 255)/256, 256, 0, stream>>>(c0, c, NPED*HD);
    copy_k<<<(NPED*2  + 255)/256, 256, 0, stream>>>(last_pos, lp, NPED*2);
    x0_k  <<<(NPED*ED + 255)/256, 256, 0, stream>>>(last_pos_rel, spd, W_sp, b_sp, x);
    prep_k<<<2, 256, 0, stream>>>(W_pe, b_pe, W_pp1, b_pp1, W3, cvec);
    bfrag_k<<<(16*2*32*8 + 255)/256, 256, 0, stream>>>(W_pp2, bfrag);

    // ---- 12 decoder steps ----
    for (int t = 0; t < PREDL; ++t) {
        int ts = (t + 1 < PREDL) ? (t + 1) : (PREDL - 1);   // spd_seq[t]
        const float* spd_t = spd + (size_t)ts * NPED;
        lstm_step<<<NPED, 256, 0, stream>>>(W_ih, b_ih, W_hh, b_hh, W_h2p, b_h2p,
                                            W_sp, b_sp, spd_t, h, c, lp, x,
                                            out + (size_t)t * NPED * 2);
        uv_kernel<<<NPED, 256, 0, stream>>>(W_pp1, W3, cvec, h, lp, spd_t, u, v);
        pool_kernel<<<NPED, 128, 0, stream>>>(u, v, bfrag, b_pp2, pool);
        mlp_kernel<<<NPED, 64, 0, stream>>>(W_m1, b_m1, W_m2, b_m2, pool, h);
    }
}
